// S4_2405181685952
// MI455X (gfx1250) — matrix operand order repaired and checked by reading, not compiled, not run on hardware
//
#include <hip/hip_runtime.h>
#include <hip/hip_bf16.h>
#include <math.h>

typedef __attribute__((ext_vector_type(16))) _Float16     v16h;
typedef __attribute__((ext_vector_type(8)))  _Float16     v8h;
typedef __attribute__((ext_vector_type(8)))  float        v8f;
typedef __attribute__((ext_vector_type(8)))  unsigned int v8u;

#define HCH   512
#define LSEQ  4096
#define NB    8
#define NFRQ  2049   // L/2 + 1
#define NMODE 32     // N/2

// LDS layout for pass 3 (bytes).  krev zero tail must cover reversed-element index
// R0 + 32*(s0+6) + 23 <= 4318, so 4352 elements is safe for all padded steps.
#define KREVA_ELEMS 4352              // krev + zero tail, copy A (shift 0)
#define KREVB_ELEMS 4368              // copy B (stored shifted by +1 element)
#define KREVA_OFF   0
#define KREVB_OFF   (KREVA_ELEMS * 2)                 // 8704
#define XT_OFF      (KREVB_OFF + KREVB_ELEMS * 2)     // 17440 (16B aligned)
#define LDS_BYTES   (XT_OFF + 16 * LSEQ * 2)          // 148512

// ---------------- complex helpers ----------------
struct c2 { float r, i; };
__device__ __forceinline__ c2 cmul(c2 a, c2 b) { return { a.r*b.r - a.i*b.i, a.r*b.i + a.i*b.r }; }
__device__ __forceinline__ c2 cadd(c2 a, c2 b) { return { a.r + b.r, a.i + b.i }; }
__device__ __forceinline__ c2 csub(c2 a, c2 b) { return { a.r - b.r, a.i - b.i }; }
__device__ __forceinline__ c2 cinv(c2 a) {
    float d = 1.f / (a.r*a.r + a.i*a.i);
    return { a.r*d, -a.i*d };
}

// ================= Pass 1: Cauchy kernel + Woodbury -> k_f[H][F] (complex f32) ============
__global__ void s4_cauchy_kernel(const float* __restrict__ log_dt,
                                 const float* __restrict__ w_ri,
                                 const float* __restrict__ B_ri,
                                 const float* __restrict__ C_ri,
                                 float* __restrict__ kf /* [H][F][2] */) {
    int idx = blockIdx.x * blockDim.x + threadIdx.x;
    if (idx >= HCH * NFRQ) return;
    int h = idx / NFRQ;
    int f = idx - h * NFRQ;

    float dt = __expf(log_dt[h]);

    float phi = -(2.0f * (float)M_PI / (float)LSEQ) * (float)f;   // freq = exp(i*phi)
    float fs, fc;
    __sincosf(phi, &fs, &fc);
    c2 onePf = { 1.f + fc, fs };
    c2 num   = { 2.f * (1.f - fc), -2.f * fs };
    c2 z     = cmul(num, cinv(onePf));                            // z = 2(1-freq)/(1+freq)

    c2 r00 = {0,0}, r01 = {0,0}, r10 = {0,0}, r11 = {0,0};

    for (int n = 0; n < NMODE; ++n) {
        float wr = w_ri[2*n], wi = w_ri[2*n + 1];
        c2 wd  = { wr * dt, wi * dt };
        c2 K1  = cinv({ z.r - wd.r, z.i - wd.i });   // 1/(z - w*dt)
        c2 K2  = cinv({ z.r - wd.r, z.i + wd.i });   // 1/(z - conj(w*dt))

        const float* Cb = C_ri + ((h * 2) * NMODE + n) * 2;
        const float* Bb = B_ri + ((h * 2) * NMODE + n) * 2;
        c2 C0 = { Cb[0], Cb[1] };
        c2 C1 = { Cb[2*NMODE + 0], Cb[2*NMODE + 1] };
        c2 B0 = { Bb[0], Bb[1] };
        c2 B1 = { Bb[2*NMODE + 0], Bb[2*NMODE + 1] };

        #define ACC(R, Ca, Bx)                                            \
        {                                                                 \
            c2 v  = { Ca.r*Bx.r + Ca.i*Bx.i, Ca.r*Bx.i - Ca.i*Bx.r };     \
            c2 vc = { v.r, -v.i };                                        \
            R = cadd(R, cadd(cmul(v, K1), cmul(vc, K2)));                 \
        }
        ACC(r00, C0, B0) ACC(r01, C0, B1) ACC(r10, C1, B0) ACC(r11, C1, B1)
        #undef ACC
    }
    r00.r *= dt; r00.i *= dt; r01.r *= dt; r01.i *= dt;
    r10.r *= dt; r10.i *= dt; r11.r *= dt; r11.i *= dt;

    c2 den = { 1.f + r11.r, r11.i };
    c2 kfc = csub(r00, cmul(cmul(r01, r10), cinv(den)));   // Woodbury
    c2 sc  = cinv(onePf); sc.r *= 2.f; sc.i *= 2.f;        // * 2/(1+freq)
    kfc = cmul(kfc, sc);

    kf[idx * 2 + 0] = kfc.r;
    kf[idx * 2 + 1] = kfc.i;
}

// ================= Pass 2: irfft via rotation recurrence -> k[H][L] as f16 ================
__global__ void s4_idft_kernel(const float* __restrict__ kf,
                               _Float16* __restrict__ kt /* [H][L] */) {
    __shared__ float kfl[NFRQ * 2];
    int h = blockIdx.y;
    int tid = threadIdx.x;
    for (int i = tid; i < NFRQ * 2; i += blockDim.x)
        kfl[i] = kf[h * NFRQ * 2 + i];
    __syncthreads();

    int l = blockIdx.x * blockDim.x + tid;
    float phi = (2.0f * (float)M_PI / (float)LSEQ) * (float)l;
    float s1, c1;
    __sincosf(phi, &s1, &c1);

    float sum = kfl[0];                                   // f = 0
    sum += ((l & 1) ? -1.f : 1.f) * kfl[2 * (NFRQ - 1)];  // f = L/2 (Nyquist)

    float c = 1.f, s = 0.f;
    for (int f = 1; f <= NFRQ - 2; ++f) {
        float cn = c * c1 - s * s1;
        float sn = s * c1 + c * s1;
        c = cn; s = sn;
        if ((f & 255) == 0) {                 // periodic exact resync
            __sincosf(phi * (float)f, &s, &c);
        }
        sum += 2.f * (kfl[2*f] * c - kfl[2*f + 1] * s);
    }
    kt[h * LSEQ + l] = (_Float16)(sum * (1.f / (float)LSEQ));
}

// ========== Pass 3: causal depthwise conv as triangular Toeplitz GEMM (WMMA f16) ==========
// One workgroup per channel h.  LDS: two 1-element-shifted copies of reversed k (zero tail
// absorbs causal masking AND tile retirement) + X_h^T [16 cols][4096] f16 (cols 8..15 zero).
// Each wave owns groups of 4 t-tiles spaced 32 apart; fragment F(s) (delta = 16*base-32s)
// lives in slot s&3, and step s computes acc_g += F(s-g)*B(s).  The 4x-unrolled body makes
// all slot indices literal -> no register shifting, no masks, no tail code.

__device__ __forceinline__ v16h load_afrag(const _Float16* p) {
    const unsigned int* p32 = (const unsigned int*)p;
    v8u t;
    t[0] = p32[0]; t[1] = p32[1]; t[2] = p32[2];  t[3] = p32[3];
    t[4] = p32[8]; t[5] = p32[9]; t[6] = p32[10]; t[7] = p32[11];
    return __builtin_bit_cast(v16h, t);
}

__device__ __forceinline__ v16h load_bfrag(const _Float16* bsrc) {
    v8h b0 = *(const v8h*)(bsrc);
    v8h b1 = *(const v8h*)(bsrc + 16);
    return __builtin_shufflevector(b0, b1, 0,1,2,3,4,5,6,7,8,9,10,11,12,13,14,15);
}

#define WMMA(A, B, C) __builtin_amdgcn_wmma_f32_16x16x32_f16(false, A, false, B, (short)0, C, false, false)

__global__ __launch_bounds__(256, 2)
void s4_conv_kernel(const float* __restrict__ x,
                    const _Float16* __restrict__ kt,
                    const float* __restrict__ D,
                    float* __restrict__ out) {
    extern __shared__ __align__(16) char smem[];
    _Float16* krevA = (_Float16*)(smem + KREVA_OFF);   // krev[i] at index i
    _Float16* krevB = (_Float16*)(smem + KREVB_OFF);   // krev[i] at index i+1
    _Float16* xT    = (_Float16*)(smem + XT_OFF);      // [16][4096], cols 8..15 zero

    const int h   = blockIdx.x;
    const int tid = threadIdx.x;

    // Stage reversed k (zero tail) twice, and X_h (transposed, f16); one HBM read of x.
    for (int i = tid; i < KREVA_ELEMS; i += 256) {
        _Float16 v = (i < LSEQ) ? kt[h * LSEQ + (LSEQ - 1 - i)] : (_Float16)0.f;
        krevA[i]     = v;
        krevB[i + 1] = v;
    }
    if (tid == 0) krevB[0] = (_Float16)0.f;
    for (int i = tid; i < NB * LSEQ; i += 256)        // i = b*4096 + t
        xT[i] = (_Float16)x[(size_t)i * HCH + h];
    for (int i = tid; i < NB * LSEQ; i += 256)        // zero pad columns 8..15
        xT[NB * LSEQ + i] = (_Float16)0.f;
    __syncthreads();

    const int wave = tid >> 5;
    const int lane = tid & 31;
    const int mrow = lane & 15;
    const int half = lane >> 4;
    const int ncol = lane & 15;
    const float Dv = D[h];

    // 64 groups of 4 t-tiles: group gi covers tt = base, base+2, base+4, base+6
    for (int gi = wave; gi < 64; gi += 8) {
        const int base = (gi >> 1) * 8 + (gi & 1);
        const int s0   = base >> 1;            // F(s) == 0 for s > s0 (zero tail)

        // reversed-element base index for this lane (parity is loop-invariant)
        const int R0 = (LSEQ - 1) - 16 * base - mrow + (half << 3);
        const _Float16* pe = (R0 & 1) ? (krevB + R0 + 1) : (krevA + R0);
        const _Float16* colp = xT + ncol * LSEQ + (half << 3);

        v8f acc0 = {0,0,0,0,0,0,0,0}, acc1 = {0,0,0,0,0,0,0,0};
        v8f acc2 = {0,0,0,0,0,0,0,0}, acc3 = {0,0,0,0,0,0,0,0};

        // prologue fragments F(-3), F(-2), F(-1) into slots 1,2,3
        v16h f1 = load_afrag(pe - 96);
        v16h f2 = load_afrag(pe - 64);
        v16h f3 = load_afrag(pe - 32);
        v16h f0;
        v16h bfr;

        // steps 0 .. STEPS-1 cover s = 0..s0+3 (real work) + <=3 zero-padded steps
        const int STEPS = (s0 + 7) & ~3;
        for (int s = 0; s < STEPS; s += 4) {
            f0  = load_afrag(pe + 32 * s);
            bfr = load_bfrag(colp + 32 * s);
            acc0 = WMMA(f0, bfr, acc0);
            acc1 = WMMA(f3, bfr, acc1);
            acc2 = WMMA(f2, bfr, acc2);
            acc3 = WMMA(f1, bfr, acc3);

            f1  = load_afrag(pe + 32 * (s + 1));
            bfr = load_bfrag(colp + 32 * (s + 1));
            acc0 = WMMA(f1, bfr, acc0);
            acc1 = WMMA(f0, bfr, acc1);
            acc2 = WMMA(f3, bfr, acc2);
            acc3 = WMMA(f2, bfr, acc3);

            f2  = load_afrag(pe + 32 * (s + 2));
            bfr = load_bfrag(colp + 32 * (s + 2));
            acc0 = WMMA(f2, bfr, acc0);
            acc1 = WMMA(f1, bfr, acc1);
            acc2 = WMMA(f0, bfr, acc2);
            acc3 = WMMA(f3, bfr, acc3);

            f3  = load_afrag(pe + 32 * (s + 3));
            bfr = load_bfrag(colp + 32 * (s + 3));
            acc0 = WMMA(f3, bfr, acc0);
            acc1 = WMMA(f2, bfr, acc1);
            acc2 = WMMA(f1, bfr, acc2);
            acc3 = WMMA(f0, bfr, acc3);
        }

        // ---- write out: C/D layout VGPR r -> M = r + 8*half, N = lane&15 ----
        if (ncol < 8) {
            const int b = ncol;
            #define STORE_TILE(ACC, G)                                               \
            {                                                                        \
                const int tb = 16 * (base + 2*(G));                                  \
                _Pragma("unroll")                                                    \
                for (int r = 0; r < 8; ++r) {                                        \
                    const int t = tb + (half << 3) + r;                              \
                    const float xv = (float)xT[b * LSEQ + t];                        \
                    out[(size_t)(b * LSEQ + t) * HCH + h] = ACC[r] + Dv * xv;        \
                }                                                                    \
            }
            STORE_TILE(acc0, 0) STORE_TILE(acc1, 1) STORE_TILE(acc2, 2) STORE_TILE(acc3, 3)
            #undef STORE_TILE
        }
    }
}

// =========================================================================================
extern "C" void kernel_launch(void* const* d_in, const int* in_sizes, int n_in,
                              void* d_out, int out_size, void* d_ws, size_t ws_size,
                              hipStream_t stream) {
    const float* x      = (const float*)d_in[0];
    const float* log_dt = (const float*)d_in[1];
    const float* w_ri   = (const float*)d_in[2];
    const float* B_ri   = (const float*)d_in[3];
    const float* C_ri   = (const float*)d_in[4];
    const float* D      = (const float*)d_in[5];
    float* out = (float*)d_out;

    // Workspace layout: kf (H*F complex f32 = 8,392,704 B) then k f16 (H*L*2 = 4 MB).
    const size_t KF_BYTES = (size_t)HCH * NFRQ * 2 * sizeof(float);
    float*    kf = (float*)d_ws;
    _Float16* kt = (_Float16*)((char*)d_ws + KF_BYTES);

    // Pass 1: Cauchy + Woodbury -> k_f
    {
        int total = HCH * NFRQ;
        s4_cauchy_kernel<<<(total + 255) / 256, 256, 0, stream>>>(log_dt, w_ri, B_ri, C_ri, kf);
    }
    // Pass 2: irfft -> k (f16)
    {
        dim3 grid(LSEQ / 256, HCH);
        s4_idft_kernel<<<grid, 256, 0, stream>>>(kf, kt);
    }
    // Pass 3: triangular Toeplitz conv with WMMA (~145 KB dynamic LDS per WG, 2 WG/WGP)
    {
        hipFuncSetAttribute((const void*)s4_conv_kernel,
                            hipFuncAttributeMaxDynamicSharedMemorySize, LDS_BYTES);
        s4_conv_kernel<<<HCH, 256, LDS_BYTES, stream>>>(x, kt, D, out);
    }
}
